// MultiHeadAttention_47158740910430
// MI455X (gfx1250) — matrix operand order repaired and checked by reading, not compiled, not run on hardware
//
#include <hip/hip_runtime.h>
#include <hip/hip_bf16.h>
#include <cstdint>

#define DM    1024
#define NH    16
#define DEPTH 64
#define BATCH 4
#define SEQ   2048

typedef _Float16 h16;
typedef __attribute__((ext_vector_type(16))) _Float16 v16h;
typedef __attribute__((ext_vector_type(8)))  _Float16 v8h;
typedef __attribute__((ext_vector_type(4)))  _Float16 v4h;
typedef __attribute__((ext_vector_type(8)))  float    v8f;
typedef __attribute__((ext_vector_type(4)))  float    v4f;
typedef __attribute__((ext_vector_type(4)))  unsigned int u32x4;
typedef __attribute__((ext_vector_type(8)))  int          i32x8;
typedef __attribute__((ext_vector_type(4)))  int          i32x4;

// LDS tile geometry: 128 rows x 64 halves, +8 halves (4 DW) row padding via
// the TDM pad feature -> row stride 72 halves (144 B) to spread the 64 banks.
#define TILE_K      64
#define TILE_MN     128
#define LDS_STRIDE  72
#define NKT         (DM / TILE_K)   // 16 K-steps

// ---------------------------------------------------------------------------
// WMMA fragment loaders (layouts per CDNA5 ISA §7.12.2, wave32):
//  A (16x32 f16): lane L -> row m=L%16; K chunks [koff,koff+8) and
//                 [16+koff,16+koff+8), koff=(L/16)*8.
//  B (32x16 f16): lane L -> col n=L%16; same K chunks as A.
//  C/D (16x16 f32): lane L -> col n=L%16; VGPR i -> row m=i+(L/16)*8.
// ---------------------------------------------------------------------------

static __device__ inline v16h frag_a_f16(const h16* __restrict__ p0, int lda, int lane) {
  const int m = lane & 15, koff = (lane >> 4) * 8;
  const h16* p = p0 + m * lda + koff;
  v8h lo = *(const v8h*)(p);
  v8h hi = *(const v8h*)(p + 16);
  return __builtin_shufflevector(lo, hi, 0,1,2,3,4,5,6,7,8,9,10,11,12,13,14,15);
}

static __device__ inline v16h frag_a_f32(const float* __restrict__ p0, int lda, int lane) {
  const int m = lane & 15, koff = (lane >> 4) * 8;
  const float* p = p0 + m * lda + koff;
  v4f x0 = *(const v4f*)(p);
  v4f x1 = *(const v4f*)(p + 4);
  v4f x2 = *(const v4f*)(p + 16);
  v4f x3 = *(const v4f*)(p + 20);
  v16h a;
#pragma unroll
  for (int i = 0; i < 4; ++i) {
    a[i]      = (_Float16)x0[i];
    a[4 + i]  = (_Float16)x1[i];
    a[8 + i]  = (_Float16)x2[i];
    a[12 + i] = (_Float16)x3[i];
  }
  return a;
}

static __device__ inline v16h frag_b_f16(const h16* __restrict__ bT, int ldt, int lane) {
  const int n = lane & 15, koff = (lane >> 4) * 8;
  const h16* p = bT + n * ldt + koff;
  v8h lo = *(const v8h*)(p);
  v8h hi = *(const v8h*)(p + 16);
  return __builtin_shufflevector(lo, hi, 0,1,2,3,4,5,6,7,8,9,10,11,12,13,14,15);
}

// LDS-resident fragments (row stride LDS_STRIDE halves, 16B aligned).
static __device__ inline v16h frag_a_lds(const h16* base, int row0, int kk, int lane) {
  const int m = lane & 15, koff = (lane >> 4) * 8;
  const h16* p = base + (row0 + m) * LDS_STRIDE + kk + koff;
  v8h lo = *(const v8h*)(p);
  v8h hi = *(const v8h*)(p + 16);
  return __builtin_shufflevector(lo, hi, 0,1,2,3,4,5,6,7,8,9,10,11,12,13,14,15);
}

static __device__ inline v16h frag_b_lds(const h16* base, int col0, int kk, int lane) {
  const int n = lane & 15, koff = (lane >> 4) * 8;
  const h16* p = base + (col0 + n) * LDS_STRIDE + kk + koff;
  v8h lo = *(const v8h*)(p);
  v8h hi = *(const v8h*)(p + 16);
  return __builtin_shufflevector(lo, hi, 0,1,2,3,4,5,6,7,8,9,10,11,12,13,14,15);
}

static __device__ inline v8f wmma_f16(v16h a, v16h b, v8f c) {
  return __builtin_amdgcn_wmma_f32_16x16x32_f16(false, a, false, b, (short)0, c, false, false);
}

// ---------------------------------------------------------------------------
// Tensor Data Mover: 2D f16 tile (tile_d1 rows x tile_d0 elems) -> LDS with
// 4-DWORD row padding.  Descriptor per CDNA5 ISA.
// Wave-uniform args.  clang-23 6-arg builtin:
//   (u32x4 g0, i32x8 g1, i32x4, i32x4, i32x8, i32 cpol); groups 2/3 unused.
// ---------------------------------------------------------------------------
static __device__ inline void tdm_load_tile_f16(const h16* gsrc, unsigned lds_off,
                                                unsigned tensor_d0, unsigned tensor_d1,
                                                unsigned tile_d0, unsigned tile_d1,
                                                unsigned row_stride_elems) {
  const unsigned long long ga = (unsigned long long)(uintptr_t)gsrc;
  u32x4 g0;
  g0[0] = 1u;                                          // count=1, user descriptor
  g0[1] = lds_off;                                     // LDS byte address
  g0[2] = (unsigned)(ga & 0xFFFFFFFFu);                // global_addr[31:0]
  g0[3] = (unsigned)((ga >> 32) & 0x01FFFFFFu)         // global_addr[56:32]
          | (2u << 30);                                // type=2 ("image")
  i32x8 g1;
  g1[0] = (int)((1u << 16)                             // data_size = 2 bytes
                | (1u << 20)                           // pad_enable
                | (4u << 22)                           // pad_interval: 32 DWORDs
                | (3u << 25));                         // pad_amount: 4 DWORDs
  g1[1] = (int)((tensor_d0 & 0xFFFFu) << 16);          // tensor_dim0[15:0]
  g1[2] = (int)((tensor_d0 >> 16) | ((tensor_d1 & 0xFFFFu) << 16));
  g1[3] = (int)((tensor_d1 >> 16) | (tile_d0 << 16));  // tensor_dim1 hi / tile_dim0
  g1[4] = (int)tile_d1;                                // tile_dim1 (tile_dim2=0)
  g1[5] = (int)row_stride_elems;                       // tensor_dim0_stride[31:0]
  g1[6] = 0;
  g1[7] = 0;
  const i32x4 zero4 = {0, 0, 0, 0};
  const i32x8 zero8 = {0, 0, 0, 0, 0, 0, 0, 0};
  __builtin_amdgcn_tensor_load_to_lds(g0, g1, zero4, zero4, zero8, 0);
}

// ---------------------------------------------------------------------------
// Kernel 0: elementwise f32 -> f16 conversion (n multiple of 4).
// ---------------------------------------------------------------------------
__global__ void __launch_bounds__(256)
cvt_f16_kernel(const float* __restrict__ src, h16* __restrict__ dst, int n) {
  const int i = (blockIdx.x * 256 + threadIdx.x) * 4;
  if (i < n) {
    v4f x = *(const v4f*)(src + i);
    v4h y;
#pragma unroll
    for (int j = 0; j < 4; ++j) y[j] = (_Float16)x[j];
    *(v4h*)(dst + i) = y;
  }
}

// ---------------------------------------------------------------------------
// Kernel 1: fused QKV projection, double-buffered TDM-tiled GEMM.
// Block = 256 threads (8 waves) -> 128x128 tile of Y = X * W^T.
// While waves compute tile t from buffer t&1, the TDM streams tile t+1 into
// the other buffer (in-order TDM retire => s_wait_tensorcnt(2) = cur done).
// z = 0/1/2 selects Q/K/V; Q pre-scaled by 1/8; V stored transposed.
// ---------------------------------------------------------------------------
__global__ void __launch_bounds__(256)
qkv_gemm_kernel(const h16* __restrict__ Xq, const h16* __restrict__ Xk,
                const h16* __restrict__ Xv,
                const h16* __restrict__ Wq, const h16* __restrict__ Wk,
                const h16* __restrict__ Wv,
                const float* __restrict__ bq, const float* __restrict__ bk,
                const float* __restrict__ bv,
                h16* __restrict__ Qh, h16* __restrict__ Kh, h16* __restrict__ Vt)
{
  __shared__ __align__(16) h16 ldsA[2][TILE_MN * LDS_STRIDE];
  __shared__ __align__(16) h16 ldsB[2][TILE_MN * LDS_STRIDE];

  const int which = blockIdx.z;
  const h16*   A    = (which == 0) ? Xq : (which == 1) ? Xk : Xv;
  const h16*   Bt   = (which == 0) ? Wq : (which == 1) ? Wk : Wv;
  const float* bias = (which == 0) ? bq : (which == 1) ? bk : bv;

  const int r0   = blockIdx.x * TILE_MN;
  const int c0   = blockIdx.y * TILE_MN;
  const int lane = threadIdx.x & 31;
  const int wv   = threadIdx.x >> 5;

  const unsigned aoff[2] = {(unsigned)(uintptr_t)(void*)ldsA[0],
                            (unsigned)(uintptr_t)(void*)ldsA[1]};
  const unsigned boff[2] = {(unsigned)(uintptr_t)(void*)ldsB[0],
                            (unsigned)(uintptr_t)(void*)ldsB[1]};
  const h16* Arow = A  + (size_t)r0 * DM;
  const h16* Brow = Bt + (size_t)c0 * DM;

  if (wv == 0) {                           // prologue: tile 0 in flight
    tdm_load_tile_f16(Arow, aoff[0], DM, BATCH * SEQ, TILE_K, TILE_MN, DM);
    tdm_load_tile_f16(Brow, boff[0], DM, DM, TILE_K, TILE_MN, DM);
  }

  v8f acc[8] = {};
  for (int t = 0; t < NKT; ++t) {
    const int cur = t & 1, nxt = cur ^ 1;
    if (wv == 0) {
      if (t + 1 < NKT) {                   // stream next tile into other buf
        const int k1 = (t + 1) * TILE_K;
        tdm_load_tile_f16(Arow + k1, aoff[nxt], DM, BATCH * SEQ, TILE_K, TILE_MN, DM);
        tdm_load_tile_f16(Brow + k1, boff[nxt], DM, DM, TILE_K, TILE_MN, DM);
        __builtin_amdgcn_s_wait_tensorcnt(2);  // current pair retired
      } else {
        __builtin_amdgcn_s_wait_tensorcnt(0);
      }
    }
    __syncthreads();                       // current tile visible to all waves
#pragma unroll
    for (int kk = 0; kk < TILE_K; kk += 32) {
      v16h a = frag_a_lds(ldsA[cur], wv * 16, kk, lane);
      v16h bf[8];
#pragma unroll
      for (int j = 0; j < 8; ++j) bf[j] = frag_b_lds(ldsB[cur], j * 16, kk, lane);
#pragma unroll
      for (int j = 0; j < 8; ++j) acc[j] = wmma_f16(a, bf[j], acc[j]);
    }
    __syncthreads();                       // done reading before overwrite
  }

  const int n  = lane & 15;
  const int mb = (lane >> 4) * 8;
  const int row_base = r0 + wv * 16 + mb;      // 8 consecutive rows from here
  const int b_idx = row_base / SEQ;
  const int s0    = row_base % SEQ;

#pragma unroll
  for (int j = 0; j < 8; ++j) {
    const int   c_n = c0 + 16 * j + n;
    const int   h   = c_n / DEPTH, d = c_n % DEPTH;
    const float bb  = bias[c_n];
    const size_t bh = (size_t)(b_idx * NH + h);
    if (which == 2) {
      v8h pk;                                  // V^T: contiguous b128 store
#pragma unroll
      for (int i = 0; i < 8; ++i) pk[i] = (h16)(acc[j][i] + bb);
      *(v8h*)&Vt[(bh * DEPTH + d) * SEQ + s0] = pk;
    } else if (which == 0) {
#pragma unroll
      for (int i = 0; i < 8; ++i)
        Qh[(bh * SEQ + (s0 + i)) * DEPTH + d] = (h16)((acc[j][i] + bb) * 0.125f);
    } else {
#pragma unroll
      for (int i = 0; i < 8; ++i)
        Kh[(bh * SEQ + (s0 + i)) * DEPTH + d] = (h16)(acc[j][i] + bb);
    }
  }
}

// ---------------------------------------------------------------------------
// Kernel 2: raw scores S = (Q/sqrt(d)) * K^T per head -> f32 in d_out.
// ---------------------------------------------------------------------------
__global__ void __launch_bounds__(32)
scores_kernel(const h16* __restrict__ Qh, const h16* __restrict__ Kh,
              float* __restrict__ attn)
{
  const int lane = threadIdx.x;
  const int q0   = blockIdx.x * 16;
  const int kv0  = blockIdx.y * 64;
  const int bh   = blockIdx.z;

  const h16* Qb = Qh + (size_t)bh * SEQ * DEPTH;
  const h16* Kb = Kh + (size_t)bh * SEQ * DEPTH;

  v8f acc[4] = {};
#pragma unroll
  for (int k0 = 0; k0 < DEPTH; k0 += 32) {
    v16h a = frag_a_f16(Qb + (size_t)q0 * DEPTH + k0, DEPTH, lane);
    v16h bf[4];
#pragma unroll
    for (int j = 0; j < 4; ++j)
      bf[j] = frag_b_f16(Kb + (size_t)(kv0 + 16 * j) * DEPTH + k0, DEPTH, lane);
#pragma unroll
    for (int j = 0; j < 4; ++j) acc[j] = wmma_f16(a, bf[j], acc[j]);
  }

  float* out = attn + (size_t)bh * SEQ * SEQ;
  const int n = lane & 15, mb = (lane >> 4) * 8;
#pragma unroll
  for (int j = 0; j < 4; ++j)
#pragma unroll
    for (int i = 0; i < 8; ++i)
      out[(size_t)(q0 + mb + i) * SEQ + (kv0 + 16 * j + n)] = acc[j][i];
}

// ---------------------------------------------------------------------------
// Kernel 3: in-place softmax over rows of 2048 (256 thr = 8 waves per row).
// ---------------------------------------------------------------------------
__global__ void __launch_bounds__(256)
softmax_kernel(float* __restrict__ attn)
{
  __shared__ float redmax[8];
  __shared__ float redsum[8];

  float* p = attn + (size_t)blockIdx.x * SEQ;
  const int t = threadIdx.x;

  float v[8];
  float mx = -3.0e38f;
#pragma unroll
  for (int i = 0; i < 8; ++i) { v[i] = p[t + 256 * i]; mx = fmaxf(mx, v[i]); }
#pragma unroll
  for (int o = 16; o > 0; o >>= 1) mx = fmaxf(mx, __shfl_xor(mx, o, 32));
  if ((t & 31) == 0) redmax[t >> 5] = mx;
  __syncthreads();
  float rowmax = redmax[0];
#pragma unroll
  for (int i = 1; i < 8; ++i) rowmax = fmaxf(rowmax, redmax[i]);

  float s = 0.f;
#pragma unroll
  for (int i = 0; i < 8; ++i) { v[i] = __expf(v[i] - rowmax); s += v[i]; }
#pragma unroll
  for (int o = 16; o > 0; o >>= 1) s += __shfl_xor(s, o, 32);
  if ((t & 31) == 0) redsum[t >> 5] = s;
  __syncthreads();
  float total = redsum[0];
#pragma unroll
  for (int i = 1; i < 8; ++i) total += redsum[i];
  const float inv = 1.0f / total;
#pragma unroll
  for (int i = 0; i < 8; ++i) p[t + 256 * i] = v[i] * inv;
}

// ---------------------------------------------------------------------------
// Kernel 4: O = P * V per head (A = weights f32 -> f16 cvt, B = V^T f16).
// ---------------------------------------------------------------------------
__global__ void __launch_bounds__(32)
attnv_kernel(const float* __restrict__ attn, const h16* __restrict__ Vt,
             h16* __restrict__ AO)
{
  const int lane = threadIdx.x;
  const int q0   = blockIdx.x * 16;
  const int bh   = blockIdx.y;

  const float* Ab = attn + (size_t)bh * SEQ * SEQ;
  const h16*   Vb = Vt   + (size_t)bh * DEPTH * SEQ;

  v8f acc[4] = {};
  for (int k0 = 0; k0 < SEQ; k0 += 32) {
    v16h a = frag_a_f32(Ab + (size_t)q0 * SEQ + k0, SEQ, lane);
    v16h bf[4];
#pragma unroll
    for (int j = 0; j < 4; ++j)
      bf[j] = frag_b_f16(Vb + (size_t)(16 * j) * SEQ + k0, SEQ, lane);
#pragma unroll
    for (int j = 0; j < 4; ++j) acc[j] = wmma_f16(a, bf[j], acc[j]);
  }

  const int n = lane & 15, mb = (lane >> 4) * 8;
  const int b_idx = bh / NH, h = bh % NH;
#pragma unroll
  for (int j = 0; j < 4; ++j)
#pragma unroll
    for (int i = 0; i < 8; ++i) {
      const int q = q0 + mb + i;
      AO[((size_t)(b_idx * SEQ + q)) * DM + h * DEPTH + 16 * j + n] =
          (h16)acc[j][i];
    }
}

// ---------------------------------------------------------------------------
// Kernel 5: output projection, same double-buffered TDM GEMM, f32 epilogue.
// ---------------------------------------------------------------------------
__global__ void __launch_bounds__(256)
out_proj_kernel(const h16* __restrict__ AO, const h16* __restrict__ Wo,
                const float* __restrict__ bo, float* __restrict__ Y)
{
  __shared__ __align__(16) h16 ldsA[2][TILE_MN * LDS_STRIDE];
  __shared__ __align__(16) h16 ldsB[2][TILE_MN * LDS_STRIDE];

  const int r0   = blockIdx.x * TILE_MN;
  const int c0   = blockIdx.y * TILE_MN;
  const int lane = threadIdx.x & 31;
  const int wv   = threadIdx.x >> 5;

  const unsigned aoff[2] = {(unsigned)(uintptr_t)(void*)ldsA[0],
                            (unsigned)(uintptr_t)(void*)ldsA[1]};
  const unsigned boff[2] = {(unsigned)(uintptr_t)(void*)ldsB[0],
                            (unsigned)(uintptr_t)(void*)ldsB[1]};
  const h16* Arow = AO + (size_t)r0 * DM;
  const h16* Brow = Wo + (size_t)c0 * DM;

  if (wv == 0) {
    tdm_load_tile_f16(Arow, aoff[0], DM, BATCH * SEQ, TILE_K, TILE_MN, DM);
    tdm_load_tile_f16(Brow, boff[0], DM, DM, TILE_K, TILE_MN, DM);
  }

  v8f acc[8] = {};
  for (int t = 0; t < NKT; ++t) {
    const int cur = t & 1, nxt = cur ^ 1;
    if (wv == 0) {
      if (t + 1 < NKT) {
        const int k1 = (t + 1) * TILE_K;
        tdm_load_tile_f16(Arow + k1, aoff[nxt], DM, BATCH * SEQ, TILE_K, TILE_MN, DM);
        tdm_load_tile_f16(Brow + k1, boff[nxt], DM, DM, TILE_K, TILE_MN, DM);
        __builtin_amdgcn_s_wait_tensorcnt(2);
      } else {
        __builtin_amdgcn_s_wait_tensorcnt(0);
      }
    }
    __syncthreads();
#pragma unroll
    for (int kk = 0; kk < TILE_K; kk += 32) {
      v16h a = frag_a_lds(ldsA[cur], wv * 16, kk, lane);
      v16h bf[8];
#pragma unroll
      for (int j = 0; j < 8; ++j) bf[j] = frag_b_lds(ldsB[cur], j * 16, kk, lane);
#pragma unroll
      for (int j = 0; j < 8; ++j) acc[j] = wmma_f16(a, bf[j], acc[j]);
    }
    __syncthreads();
  }

  const int n  = lane & 15;
  const int mb = (lane >> 4) * 8;
  const int row_base = r0 + wv * 16 + mb;
#pragma unroll
  for (int j = 0; j < 8; ++j) {
    const int c_n = c0 + 16 * j + n;
    const float bb = bo[c_n];
#pragma unroll
    for (int i = 0; i < 8; ++i)
      Y[(size_t)(row_base + i) * DM + c_n] = acc[j][i] + bb;
  }
}

// ---------------------------------------------------------------------------
extern "C" void kernel_launch(void* const* d_in, const int* in_sizes, int n_in,
                              void* d_out, int out_size, void* d_ws, size_t ws_size,
                              hipStream_t stream) {
  const float* query = (const float*)d_in[0];
  const float* key_  = (const float*)d_in[1];
  const float* value = (const float*)d_in[2];
  const float* Wq    = (const float*)d_in[3];
  const float* bq    = (const float*)d_in[4];
  const float* Wk    = (const float*)d_in[5];
  const float* bk    = (const float*)d_in[6];
  const float* Wv    = (const float*)d_in[7];
  const float* bv    = (const float*)d_in[8];
  const float* Wo    = (const float*)d_in[9];
  const float* bo    = (const float*)d_in[10];

  const size_t MB = 1048576;
  char* ws = (char*)d_ws;
  h16* Xq16 = (h16*)(ws +   0 * MB);   // [B*S, DM] f16
  h16* Xk16 = (h16*)(ws +  16 * MB);
  h16* Xv16 = (h16*)(ws +  32 * MB);
  h16* Wq16 = (h16*)(ws +  48 * MB);   // [DM, DM] f16
  h16* Wk16 = (h16*)(ws +  50 * MB);
  h16* Wv16 = (h16*)(ws +  52 * MB);
  h16* Wo16 = (h16*)(ws +  54 * MB);
  h16* Qh   = (h16*)(ws +  56 * MB);   // [B*H, S, 64] (pre-scaled)
  h16* Kh   = (h16*)(ws +  72 * MB);   // [B*H, S, 64]
  h16* Vt   = (h16*)(ws +  88 * MB);   // [B*H, 64, S] (transposed)
  h16* AO   = (h16*)(ws + 104 * MB);   // [B, S, DM]   (head-merged)

  float* out  = (float*)d_out;                   // [B, S, DM]
  float* attn = out + (size_t)BATCH * SEQ * DM;  // [B, H, S, S]

  const int NX = BATCH * SEQ * DM;  // 8,388,608
  const int NW = DM * DM;           // 1,048,576
  cvt_f16_kernel<<<NX / 1024, 256, 0, stream>>>(query, Xq16, NX);
  cvt_f16_kernel<<<NX / 1024, 256, 0, stream>>>(key_,  Xk16, NX);
  cvt_f16_kernel<<<NX / 1024, 256, 0, stream>>>(value, Xv16, NX);
  cvt_f16_kernel<<<NW / 1024, 256, 0, stream>>>(Wq, Wq16, NW);
  cvt_f16_kernel<<<NW / 1024, 256, 0, stream>>>(Wk, Wk16, NW);
  cvt_f16_kernel<<<NW / 1024, 256, 0, stream>>>(Wv, Wv16, NW);
  cvt_f16_kernel<<<NW / 1024, 256, 0, stream>>>(Wo, Wo16, NW);

  qkv_gemm_kernel<<<dim3((BATCH * SEQ) / TILE_MN, DM / TILE_MN, 3),
                    dim3(256), 0, stream>>>(
      Xq16, Xk16, Xv16, Wq16, Wk16, Wv16, bq, bk, bv, Qh, Kh, Vt);

  scores_kernel<<<dim3(SEQ / 16, SEQ / 64, BATCH * NH), dim3(32), 0, stream>>>(
      Qh, Kh, attn);

  softmax_kernel<<<dim3(BATCH * NH * SEQ), dim3(256), 0, stream>>>(attn);

  attnv_kernel<<<dim3(SEQ / 16, BATCH * NH), dim3(32), 0, stream>>>(attn, Vt, AO);

  out_proj_kernel<<<dim3((BATCH * SEQ) / TILE_MN, DM / TILE_MN),
                    dim3(256), 0, stream>>>(AO, Wo16, bo, out);
}
